// DecoderBlock_63273458205166
// MI455X (gfx1250) — compile-verified, matrix operand order checked by reading, not run on hardware
//
#include <hip/hip_runtime.h>
#include <hip/hip_bf16.h>

// ---------------------------------------------------------------------------
// Decoder block for MI455X (gfx1250): all GEMMs + attention on
// v_wmma_f32_16x16x32_bf16 (wave32, fp32 accumulate), with double-buffered
// async global->LDS staging (ASYNCcnt) when the toolchain exposes it.
// ---------------------------------------------------------------------------

#define B_SZ    2
#define S_TOK   2048
#define DMODEL  1024
#define DFF_    4096
#define H_      16
#define M_ROWS  (B_SZ * S_TOK)   // 4096

typedef unsigned short u16;
typedef unsigned int   u32;
typedef __attribute__((ext_vector_type(16))) __bf16 v16bf;
typedef __attribute__((ext_vector_type(8)))  float  v8f;

union Frag16 {            // one 16x32 bf16 WMMA operand (8 VGPRs / lane)
    v16bf v;
    uint4 u[2];
};

__device__ __forceinline__ u16 f2bf(float f) {
    u32 u = __builtin_bit_cast(u32, f);
    u += 0x7fffu + ((u >> 16) & 1u);     // round-to-nearest-even
    return (u16)(u >> 16);
}

// ---- async global->LDS (16B per lane), with synchronous fallback ----------
#if defined(__gfx1250__) && __has_builtin(__builtin_amdgcn_global_load_async_to_lds_b128)
#define HAVE_ASYNC 1
#else
#define HAVE_ASYNC 0
#endif

typedef int vec4i __attribute__((vector_size(16)));
typedef __attribute__((address_space(1))) vec4i as1_v4i;
typedef __attribute__((address_space(3))) vec4i as3_v4i;

__device__ __forceinline__ void async_cp16(const u16* g, u16* l) {
#if HAVE_ASYNC
    vec4i* gg = (vec4i*)const_cast<u16*>(g);   // pointee change, generic AS
    vec4i* ll = (vec4i*)l;
    __builtin_amdgcn_global_load_async_to_lds_b128(
        (as1_v4i*)gg, (as3_v4i*)ll, 0, 0);
#else
    *(uint4*)l = *(const uint4*)g;
#endif
}

__device__ __forceinline__ void async_wait() {
#if HAVE_ASYNC
#if __has_builtin(__builtin_amdgcn_s_wait_asynccnt)
    __builtin_amdgcn_s_wait_asynccnt(0);
#else
    asm volatile("s_wait_asynccnt 0x0" ::: "memory");
#endif
#endif
}

// ---------------------------------------------------------------------------
// Weight transpose + fp32 -> bf16 convert:  Wt[n*K + k] = bf16(W[k*N + n])
// ---------------------------------------------------------------------------
__global__ __launch_bounds__(256) void tconv_kernel(const float* __restrict__ W,
                                                    u16* __restrict__ Wt,
                                                    int K, int N) {
    __shared__ float tile[32][33];
    int tx = threadIdx.x, ty = threadIdx.y;
    int n0 = blockIdx.x * 32, k0 = blockIdx.y * 32;
#pragma unroll
    for (int j = 0; j < 4; ++j)
        tile[ty + j * 8][tx] = W[(size_t)(k0 + ty + j * 8) * N + n0 + tx];
    __syncthreads();
#pragma unroll
    for (int j = 0; j < 4; ++j)
        Wt[(size_t)(n0 + ty + j * 8) * K + k0 + tx] = f2bf(tile[tx][ty + j * 8]);
}

// ---------------------------------------------------------------------------
// Per-head V transpose: V[b, s, h*64+d] (bf16) -> VT[((b*H+h)*64+d)*S + s]
// so the attention hot loop can async-load V^T tiles contiguously.
// ---------------------------------------------------------------------------
__global__ __launch_bounds__(256) void vtrans_kernel(const u16* __restrict__ V,
                                                     u16* __restrict__ VT) {
    __shared__ u16 tile[32][33];
    int tx = threadIdx.x, ty = threadIdx.y;
    int s0 = blockIdx.x * 32, d0 = blockIdx.y * 32;
    int bh = blockIdx.z, b = bh >> 4, h = bh & 15;
#pragma unroll
    for (int j = 0; j < 4; ++j)
        tile[ty + j * 8][tx] =
            V[((size_t)b * S_TOK + s0 + ty + j * 8) * DMODEL + h * 64 + d0 + tx];
    __syncthreads();
#pragma unroll
    for (int j = 0; j < 4; ++j)
        VT[((size_t)bh * 64 + d0 + ty + j * 8) * S_TOK + s0 + tx] =
            tile[tx][ty + j * 8];
}

// ---------------------------------------------------------------------------
// LayerNorm over D=1024; writes fp32 (optional, residual path) + bf16 copy.
// ---------------------------------------------------------------------------
__global__ __launch_bounds__(256) void ln_kernel(const float* __restrict__ X,
                                                 const float* __restrict__ gw,
                                                 const float* __restrict__ bw,
                                                 float* __restrict__ Hf,
                                                 u16* __restrict__ Hb) {
    __shared__ float red_s[8], red_q[8];
    int tid = threadIdx.x, lane = tid & 31, w = tid >> 5;
    size_t base = (size_t)blockIdx.x * DMODEL;
    int idx = tid * 4;

    float4 xv = *(const float4*)&X[base + idx];
    float s  = xv.x + xv.y + xv.z + xv.w;
    float ss = xv.x * xv.x + xv.y * xv.y + xv.z * xv.z + xv.w * xv.w;
#pragma unroll
    for (int off = 16; off >= 1; off >>= 1) {
        s  += __shfl_xor(s, off);
        ss += __shfl_xor(ss, off);
    }
    if (lane == 0) { red_s[w] = s; red_q[w] = ss; }
    __syncthreads();
    if (tid == 0) {
        float a = 0.f, b = 0.f;
#pragma unroll
        for (int i = 0; i < 8; ++i) { a += red_s[i]; b += red_q[i]; }
        red_s[0] = a; red_q[0] = b;
    }
    __syncthreads();
    float mean = red_s[0] * (1.0f / DMODEL);
    float var  = red_q[0] * (1.0f / DMODEL) - mean * mean;
    float rs   = rsqrtf(var + 1e-5f);

    float4 gv = *(const float4*)&gw[idx];
    float4 bv = *(const float4*)&bw[idx];
    float y0 = (xv.x - mean) * rs * gv.x + bv.x;
    float y1 = (xv.y - mean) * rs * gv.y + bv.y;
    float y2 = (xv.z - mean) * rs * gv.z + bv.z;
    float y3 = (xv.w - mean) * rs * gv.w + bv.w;
    if (Hf) {
        float4 yv = {y0, y1, y2, y3};
        *(float4*)&Hf[base + idx] = yv;
    }
    uint2 pk;
    pk.x = (u32)f2bf(y0) | ((u32)f2bf(y1) << 16);
    pk.y = (u32)f2bf(y2) | ((u32)f2bf(y3) << 16);
    *(uint2*)&Hb[base + idx] = pk;
}

// ---------------------------------------------------------------------------
// bf16 WMMA GEMM: C[M,N] = A[M,K] (row-major bf16) x Bt[N,K]^T + bias
// 128x128 block tile, 8 waves of 32x64, K staged 32 wide through LDS with a
// double-buffered async global->LDS pipeline.
// ---------------------------------------------------------------------------
template <bool RELU, bool RESID, bool OUTBF>
__global__ __launch_bounds__(256) void gemm_bf16(const u16* __restrict__ A,
                                                 const u16* __restrict__ Bt,
                                                 const float* __restrict__ bias,
                                                 const float* __restrict__ resid,
                                                 float* __restrict__ Cf,
                                                 u16* __restrict__ Cb,
                                                 int M, int N, int K) {
    __shared__ u16 As[2][128 * 32];
    __shared__ u16 Bs[2][128 * 32];

    int tid = threadIdx.x, lane = tid & 31, w = tid >> 5;
    int ln = lane & 15, lh = lane >> 4;
    int wr = w >> 1, wc = w & 1;               // 4x2 wave grid
    int m0 = blockIdx.y * 128, n0 = blockIdx.x * 128;

    v8f acc[2][4];
#pragma unroll
    for (int f = 0; f < 2; ++f)
#pragma unroll
        for (int g = 0; g < 4; ++g)
#pragma unroll
            for (int r = 0; r < 8; ++r) acc[f][g][r] = 0.0f;

    auto stage = [&](int buf, int kk) {
#pragma unroll
        for (int i = 0; i < 2; ++i) {
            int idx = tid + i * 256;
            int row = idx >> 2, c = idx & 3;
            async_cp16(&A[(size_t)(m0 + row) * K + kk + c * 8],
                       &As[buf][row * 32 + c * 8]);
            async_cp16(&Bt[(size_t)(n0 + row) * K + kk + c * 8],
                       &Bs[buf][row * 32 + c * 8]);
        }
    };

    int T = K / 32;
    stage(0, 0);
    async_wait();
    __syncthreads();

    for (int t = 0; t < T; ++t) {
        int cur = t & 1;
        if (t + 1 < T) stage(1 - cur, (t + 1) * 32);   // overlap with WMMAs

        Frag16 aF[2], bF[4];
#pragma unroll
        for (int f = 0; f < 2; ++f) {
            int row = wr * 32 + f * 16 + ln;
            aF[f].u[0] = *(const uint4*)&As[cur][row * 32 + lh * 8];
            aF[f].u[1] = *(const uint4*)&As[cur][row * 32 + 16 + lh * 8];
        }
#pragma unroll
        for (int g = 0; g < 4; ++g) {
            int row = wc * 64 + g * 16 + ln;
            bF[g].u[0] = *(const uint4*)&Bs[cur][row * 32 + lh * 8];
            bF[g].u[1] = *(const uint4*)&Bs[cur][row * 32 + 16 + lh * 8];
        }
#pragma unroll
        for (int f = 0; f < 2; ++f)
#pragma unroll
            for (int g = 0; g < 4; ++g)
                acc[f][g] = __builtin_amdgcn_wmma_f32_16x16x32_bf16(
                    false, aF[f].v, false, bF[g].v, (short)0, acc[f][g],
                    false, false);

        async_wait();
        __syncthreads();
    }

#pragma unroll
    for (int f = 0; f < 2; ++f)
#pragma unroll
        for (int g = 0; g < 4; ++g)
#pragma unroll
            for (int r = 0; r < 8; ++r) {
                int row = m0 + wr * 32 + f * 16 + r + 8 * lh;
                int col = n0 + wc * 64 + g * 16 + ln;
                float v = acc[f][g][r] + bias[col];
                if (RESID) v += resid[(size_t)row * N + col];
                if (RELU)  v = fmaxf(v, 0.0f);
                if (OUTBF) Cb[(size_t)row * N + col] = f2bf(v);
                else       Cf[(size_t)row * N + col] = v;
            }
}

// ---------------------------------------------------------------------------
// Attention with the reference's triu/transpose masking trick:
//   logit[q,k] = (k>=q) ? s/sqrt(D) : 0      (goes into the softmax sum)
//   weight[q,k] = (k<=q) ? softmax(logit)[k] : 0
// One block = (b,h) x 64 q rows; 4 wave32s each own 16 q rows.
// K and V^T tiles double-buffered through async global->LDS.
// ---------------------------------------------------------------------------
__global__ __launch_bounds__(128) void attn_kernel(const u16* __restrict__ Qg,
                                                   const u16* __restrict__ Kg,
                                                   const u16* __restrict__ VTg,
                                                   u16* __restrict__ Og) {
    __shared__ u16 Ks[2][32 * 64];    // [key][d]
    __shared__ u16 Vt[2][64 * 32];    // [d][key]  (already transposed in global)
    __shared__ u16 Ps[4][16 * 32];    // per-wave P tile, A-fragment staging

    int tid = threadIdx.x, lane = tid & 31, w = tid >> 5;
    int ln = lane & 15, lh = lane >> 4;
    int bh = blockIdx.y, b = bh >> 4, h = bh & 15;
    int q0 = blockIdx.x * 64 + w * 16;
    size_t hbase  = (size_t)b * S_TOK * DMODEL + h * 64;
    size_t vtbase = (size_t)bh * 64 * S_TOK;

    // Q fragments (held in VGPRs for the whole kernel)
    Frag16 qa[2];
#pragma unroll
    for (int s = 0; s < 2; ++s) {
        size_t a = hbase + (size_t)(q0 + ln) * DMODEL + s * 32 + lh * 8;
        qa[s].u[0] = *(const uint4*)&Qg[a];
        qa[s].u[1] = *(const uint4*)&Qg[a + 16];
    }

    float m[8], l[8];
    v8f o[4];
#pragma unroll
    for (int r = 0; r < 8; ++r) { m[r] = -1e30f; l[r] = 0.0f; }
#pragma unroll
    for (int g = 0; g < 4; ++g)
#pragma unroll
        for (int r = 0; r < 8; ++r) o[g][r] = 0.0f;

    const float scale = 0.03125f;     // 1/sqrt(1024)

    auto stageKV = [&](int buf, int kt) {
        int kb0 = kt * 32;
#pragma unroll
        for (int i = 0; i < 2; ++i) {
            int idx = tid + i * 128;
            int kr = idx >> 3, c = idx & 7;        // K: 32 rows x 64 d
            async_cp16(&Kg[hbase + (size_t)(kb0 + kr) * DMODEL + c * 8],
                       &Ks[buf][kr * 64 + c * 8]);
            int vr = idx >> 2, vc = idx & 3;       // V^T: 64 d rows x 32 keys
            async_cp16(&VTg[vtbase + (size_t)vr * S_TOK + kb0 + vc * 8],
                       &Vt[buf][vr * 32 + vc * 8]);
        }
    };

    const int T = S_TOK / 32;
    stageKV(0, 0);
    async_wait();
    __syncthreads();

    for (int kt = 0; kt < T; ++kt) {
        int cur = kt & 1;
        int kb0 = kt * 32;
        if (kt + 1 < T) stageKV(1 - cur, kt + 1);

        // scores: 16q x 32k via 2 key-blocks x 2 d-steps of WMMA
        v8f cs[2];
#pragma unroll
        for (int kb = 0; kb < 2; ++kb) {
            v8f c;
#pragma unroll
            for (int r = 0; r < 8; ++r) c[r] = 0.0f;
#pragma unroll
            for (int s = 0; s < 2; ++s) {
                Frag16 bF;
                int rowN = kb * 16 + ln;
                bF.u[0] = *(const uint4*)&Ks[cur][rowN * 64 + s * 32 + lh * 8];
                bF.u[1] = *(const uint4*)&Ks[cur][rowN * 64 + s * 32 + 16 + lh * 8];
                c = __builtin_amdgcn_wmma_f32_16x16x32_bf16(
                    false, qa[s].v, false, bF.v, (short)0, c, false, false);
            }
            cs[kb] = c;
        }

        // online softmax with the reference's zero-masking semantics
#pragma unroll
        for (int r = 0; r < 8; ++r) {
            int rowq = q0 + r + 8 * lh;
            int key0 = kb0 + ln, key1 = kb0 + 16 + ln;
            float lg0 = (key0 >= rowq) ? cs[0][r] * scale : 0.0f;
            float lg1 = (key1 >= rowq) ? cs[1][r] * scale : 0.0f;
            float mx = fmaxf(lg0, lg1);
#pragma unroll
            for (int off = 8; off >= 1; off >>= 1) mx = fmaxf(mx, __shfl_xor(mx, off));
            float mn   = fmaxf(m[r], mx);
            float ascl = __expf(m[r] - mn);
            m[r] = mn;
            float e0 = __expf(lg0 - mn), e1 = __expf(lg1 - mn);
            float rs = e0 + e1;
#pragma unroll
            for (int off = 8; off >= 1; off >>= 1) rs += __shfl_xor(rs, off);
            l[r] = l[r] * ascl + rs;
#pragma unroll
            for (int g = 0; g < 4; ++g) o[g][r] *= ascl;
            float w0 = (key0 <= rowq) ? e0 : 0.0f;   // numerator keeps k<=q only
            float w1 = (key1 <= rowq) ? e1 : 0.0f;
            Ps[w][(r + 8 * lh) * 32 + ln]      = f2bf(w0);
            Ps[w][(r + 8 * lh) * 32 + 16 + ln] = f2bf(w1);
        }
        // Ps is a per-wave region; same-wave LDS ops are in-order (DScnt),
        // so no workgroup barrier is needed between the stores and loads.

        // O += P x V  (P from LDS in A-fragment layout, V^T from LDS)
        Frag16 pa;
        pa.u[0] = *(const uint4*)&Ps[w][ln * 32 + lh * 8];
        pa.u[1] = *(const uint4*)&Ps[w][ln * 32 + 16 + lh * 8];
#pragma unroll
        for (int g = 0; g < 4; ++g) {
            Frag16 bV;
            int rowN = g * 16 + ln;
            bV.u[0] = *(const uint4*)&Vt[cur][rowN * 32 + lh * 8];
            bV.u[1] = *(const uint4*)&Vt[cur][rowN * 32 + 16 + lh * 8];
            o[g] = __builtin_amdgcn_wmma_f32_16x16x32_bf16(
                false, pa.v, false, bV.v, (short)0, o[g], false, false);
        }

        async_wait();
        __syncthreads();
    }

#pragma unroll
    for (int g = 0; g < 4; ++g)
#pragma unroll
        for (int r = 0; r < 8; ++r) {
            int rowq = q0 + r + 8 * lh;
            int col  = h * 64 + g * 16 + ln;
            float v = o[g][r] / l[r];
            Og[((size_t)b * S_TOK + rowq) * DMODEL + col] = f2bf(v);
        }
}

// ---------------------------------------------------------------------------
extern "C" void kernel_launch(void* const* d_in, const int* in_sizes, int n_in,
                              void* d_out, int out_size, void* d_ws, size_t ws_size,
                              hipStream_t stream) {
    (void)in_sizes; (void)n_in; (void)out_size; (void)ws_size;
    const float* x   = (const float*)d_in[0];
    const float* g1  = (const float*)d_in[1];
    const float* be1 = (const float*)d_in[2];
    const float* wq  = (const float*)d_in[3];
    const float* bq  = (const float*)d_in[4];
    const float* wk  = (const float*)d_in[5];
    const float* bk  = (const float*)d_in[6];
    const float* wv  = (const float*)d_in[7];
    const float* bv  = (const float*)d_in[8];
    const float* wo  = (const float*)d_in[9];
    const float* bo  = (const float*)d_in[10];
    const float* g2  = (const float*)d_in[11];
    const float* be2 = (const float*)d_in[12];
    const float* w1  = (const float*)d_in[13];
    const float* b1  = (const float*)d_in[14];
    const float* w2  = (const float*)d_in[15];
    const float* b2  = (const float*)d_in[16];

    char* ws = (char*)d_ws;
    const size_t MB = 1ull << 20;
    float* h_f32   = (float*)(ws + 0);        // 16 MB  LN1 out (residual)
    u16*   h_bf    = (u16*)(ws + 16 * MB);    //  8 MB  LN1 out bf16
    u16*   q_bf    = (u16*)(ws + 24 * MB);    //  8 MB
    u16*   k_bf    = (u16*)(ws + 32 * MB);    //  8 MB
    u16*   v_bf    = (u16*)(ws + 40 * MB);    //  8 MB
    u16*   attn_bf = (u16*)(ws + 48 * MB);    //  8 MB
    float* h2_f32  = (float*)(ws + 56 * MB);  // 16 MB  residual sum
    u16*   h2_bf   = (u16*)(ws + 72 * MB);    //  8 MB  LN2 out bf16
    u16*   wqT     = (u16*)(ws + 80 * MB);
    u16*   wkT     = (u16*)(ws + 82 * MB);
    u16*   wvT     = (u16*)(ws + 84 * MB);
    u16*   woT     = (u16*)(ws + 86 * MB);
    u16*   w1T     = (u16*)(ws + 88 * MB);    //  8 MB
    u16*   w2T     = (u16*)(ws + 96 * MB);    //  8 MB
    u16*   vT      = (u16*)(ws + 104 * MB);   //  8 MB per-head V^T (ends 112 MB)
    u16*   f1_bf   = q_bf;                    // reuse 24..56 MB for FFN1 acts

    dim3 tcb(32, 8);
    tconv_kernel<<<dim3(DMODEL / 32, DMODEL / 32), tcb, 0, stream>>>(wq, wqT, DMODEL, DMODEL);
    tconv_kernel<<<dim3(DMODEL / 32, DMODEL / 32), tcb, 0, stream>>>(wk, wkT, DMODEL, DMODEL);
    tconv_kernel<<<dim3(DMODEL / 32, DMODEL / 32), tcb, 0, stream>>>(wv, wvT, DMODEL, DMODEL);
    tconv_kernel<<<dim3(DMODEL / 32, DMODEL / 32), tcb, 0, stream>>>(wo, woT, DMODEL, DMODEL);
    tconv_kernel<<<dim3(DFF_ / 32, DMODEL / 32), tcb, 0, stream>>>(w1, w1T, DMODEL, DFF_);
    tconv_kernel<<<dim3(DMODEL / 32, DFF_ / 32), tcb, 0, stream>>>(w2, w2T, DFF_, DMODEL);

    ln_kernel<<<M_ROWS, 256, 0, stream>>>(x, g1, be1, h_f32, h_bf);

    dim3 gq(DMODEL / 128, M_ROWS / 128);
    gemm_bf16<false, false, true><<<gq, 256, 0, stream>>>(h_bf, wqT, bq, nullptr, nullptr, q_bf, M_ROWS, DMODEL, DMODEL);
    gemm_bf16<false, false, true><<<gq, 256, 0, stream>>>(h_bf, wkT, bk, nullptr, nullptr, k_bf, M_ROWS, DMODEL, DMODEL);
    gemm_bf16<false, false, true><<<gq, 256, 0, stream>>>(h_bf, wvT, bv, nullptr, nullptr, v_bf, M_ROWS, DMODEL, DMODEL);

    vtrans_kernel<<<dim3(S_TOK / 32, 2, B_SZ * H_), tcb, 0, stream>>>(v_bf, vT);

    attn_kernel<<<dim3(S_TOK / 64, B_SZ * H_), 128, 0, stream>>>(q_bf, k_bf, vT, attn_bf);

    gemm_bf16<false, true, false><<<gq, 256, 0, stream>>>(attn_bf, woT, bo, h_f32, h2_f32, nullptr, M_ROWS, DMODEL, DMODEL);

    ln_kernel<<<M_ROWS, 256, 0, stream>>>(h2_f32, g2, be2, nullptr, h2_bf);

    gemm_bf16<true, false, true><<<dim3(DFF_ / 128, M_ROWS / 128), 256, 0, stream>>>(h2_bf, w1T, b1, nullptr, nullptr, f1_bf, M_ROWS, DFF_, DMODEL);

    gemm_bf16<true, false, false><<<gq, 256, 0, stream>>>(f1_bf, w2T, b2, nullptr, (float*)d_out, nullptr, M_ROWS, DMODEL, DFF_);
}
